// RnnTagger_79869211836654
// MI455X (gfx1250) — matrix operand order repaired and checked by reading, not compiled, not run on hardware
//
#include <hip/hip_runtime.h>
#include <hip/hip_bf16.h>

// ---------------------------------------------------------------------------
// GRU tagger on MI455X (gfx1250): wave32 WMMA bf16 16x16x32, LDS-resident
// recurrent weights, per-step async-DMA staging of [x_t|h] into LDS
// (global_load_async_to_lds_b128 / ASYNCcnt), persistent 16-WG scan with a
// device-scope generation barrier.
// ---------------------------------------------------------------------------

#define EE 256
#define HH 512
#define TT 64
#define BB 64
#define SS 512
#define NBLK_SCAN 16          // 16 WGs, each owns 32 hidden units
#define JSL 32                // hidden units per WG
#define KTOT 768              // E + H concatenated K

// LDS layout (bytes) inside one scan workgroup:
#define LDS_W   0                       // 128 rows x 768 bf16   = 196608
#define LDS_A   196608                  // A panel 64x768 bf16   =  98304
#define LDS_G   196608                  // gates 64x128 f32 (overlays A)
#define LDS_BS  (196608 + 98304)        // bias 128 f32          =    512
#define LDS_H   (LDS_BS + 512)          // h slice 64x32 f32     =   8192
#define SMEM_TOTAL (LDS_H + 8192)       // 303616 < 320K

typedef __attribute__((ext_vector_type(16))) __bf16 v16bf;
typedef __attribute__((ext_vector_type(8)))  float  v8f;

union BfVec { uint4 q[2]; v16bf v; };

__device__ __forceinline__ unsigned short f2bf(float f) {
  unsigned int u = __float_as_uint(f);
  unsigned int r = (u + 0x7FFFu + ((u >> 16) & 1u)) >> 16;
  return (unsigned short)r;
}

// A tile (16x32 bf16, MxK) from row-major bf16 [rows][ld]:
// lane<16: M=lane,    elems 0..7 = K k0+0..7,  elems 8..15 = K k0+16..23
// lane>=16: M=lane-16, elems 0..7 = K k0+8..15, elems 8..15 = K k0+24..31
__device__ __forceinline__ v16bf load_a16x32(const unsigned short* p, int ld,
                                             int row0, int k0, int lane) {
  int m  = lane & 15;
  int kh = lane >> 4;
  const unsigned short* r = p + (size_t)(row0 + m) * ld + k0 + kh * 8;
  BfVec t;
  t.q[0] = *(const uint4*)(r);
  t.q[1] = *(const uint4*)(r + 16);
  return t.v;
}

// B tile (32x16 bf16, KxN), B[k][n] = W[n][k], W row-major [N][ldk]:
// lane: n = col0 + (lane&15); h = lane>>4; elems 0..7 = K k0+8h+0..7, elems 8..15 = K k0+16+8h+0..7
__device__ __forceinline__ v16bf load_b32x16(const unsigned short* w, int ldk,
                                             int col0, int k0, int lane) {
  int n  = col0 + (lane & 15);
  int kh = lane >> 4;
  const unsigned short* r = w + (size_t)n * ldk + k0 + kh * 8;
  BfVec t;
  t.q[0] = *(const uint4*)(r);
  t.q[1] = *(const uint4*)(r + 16);
  return t.v;
}

__device__ __forceinline__ v8f wmma_bf16(v16bf a, v16bf b, v8f c) {
  return __builtin_amdgcn_wmma_f32_16x16x32_bf16(false, a, false, b,
                                                 (short)0, c, false, false);
}

// Async DMA: LDS[lds_byte] = MEM[gptr], 16 bytes per lane, ASYNCcnt-tracked.
__device__ __forceinline__ void async_copy_b128(unsigned int lds_byte,
                                                const void* gptr) {
  asm volatile("global_load_async_to_lds_b128 %0, %1, off"
               :: "v"(lds_byte), "v"(gptr) : "memory");
}
__device__ __forceinline__ void wait_asynccnt0() {
  asm volatile("s_wait_asynccnt 0x0" ::: "memory");
}

// ------------------------- kernel 1: gather + bf16 -------------------------
// x_bf[s*64 + b][e] = bf16(emb[tokens[b][s]][e]), time-major rows.
__global__ void k_gather_x(const int* __restrict__ tokens,
                           const float* __restrict__ emb,
                           unsigned short* __restrict__ x_bf) {
  int idx = (blockIdx.x * blockDim.x + threadIdx.x) * 4;
  if (idx >= SS * BB * EE) return;
  int r  = idx >> 8;
  int e0 = idx & 255;
  int s  = r >> 6;
  int b  = r & 63;
  int tok = tokens[b * SS + s];
  const float* src = emb + (size_t)tok * EE + e0;
  unsigned short* dst = x_bf + (size_t)r * EE + e0;
  float4 v = *(const float4*)src;
  dst[0] = f2bf(v.x); dst[1] = f2bf(v.y); dst[2] = f2bf(v.z); dst[3] = f2bf(v.w);
}

// ------------------------- kernel 2: W_tag -> bf16 -------------------------
__global__ void k_conv_wtag(const float* __restrict__ w,
                            unsigned short* __restrict__ wb) {
  int i = blockIdx.x * blockDim.x + threadIdx.x;
  if (i < TT * HH) wb[i] = f2bf(w[i]);
}

// ------------------------- grid barrier (device scope) ---------------------
__device__ __forceinline__ void grid_sync(int* cnt, int* gen, int nblk) {
  __syncthreads();
  __threadfence();
  if (threadIdx.x == 0) {
    int g  = __hip_atomic_load(gen, __ATOMIC_RELAXED, __HIP_MEMORY_SCOPE_AGENT);
    int my = __hip_atomic_fetch_add(cnt, 1, __ATOMIC_ACQ_REL, __HIP_MEMORY_SCOPE_AGENT);
    if (my == nblk - 1) {
      __hip_atomic_store(cnt, 0, __ATOMIC_RELAXED, __HIP_MEMORY_SCOPE_AGENT);
      __hip_atomic_fetch_add(gen, 1, __ATOMIC_RELEASE, __HIP_MEMORY_SCOPE_AGENT);
    } else {
      while (__hip_atomic_load(gen, __ATOMIC_ACQUIRE, __HIP_MEMORY_SCOPE_AGENT) == g)
        __builtin_amdgcn_s_sleep(1);
    }
  }
  __syncthreads();
  __threadfence();
}

// ------------------------- kernel 3: persistent GRU scan -------------------
// 16 blocks x 256 threads, one block per WGP (297KB LDS) -> spend VGPRs.
__global__ void __launch_bounds__(256, 1)
k_gru_scan(const unsigned short* __restrict__ x_bf,
           const float* __restrict__ W_ih,
           const float* __restrict__ W_hh,
           const float* __restrict__ b_ih,
           const float* __restrict__ b_hh,
           unsigned short* __restrict__ h_bf,  // [64][512]
           unsigned short* __restrict__ st_bf, // [B*S][512]
           int* __restrict__ cnt, int* __restrict__ gen) {
  const int wg   = blockIdx.x;
  const int tid  = threadIdx.x;
  const int wave = tid >> 5;
  const int lane = tid & 31;

  extern __shared__ char smem[];
  unsigned short* Wl  = (unsigned short*)(smem + LDS_W);  // 128 x 768 bf16
  unsigned short* Als = (unsigned short*)(smem + LDS_A);  // 64 x 768 bf16
  float*          gts = (float*)(smem + LDS_G);           // 64 x 128 f32 (overlay)
  float*          bsl = (float*)(smem + LDS_BS);          // 128 f32
  float*          hsl = (float*)(smem + LDS_H);           // 64 x 32 f32
  const unsigned int als_base = (unsigned int)(uintptr_t)(void*)Als; // LDS byte off

  // ---- stage weights into LDS (f32 -> bf16); groups: r, z, n_i, n_h ----
  for (int i = tid; i < 128 * KTOT; i += blockDim.x) {
    int c = i / KTOT, k = i - c * KTOT;
    int grp = c >> 5, jj = c & 31;
    int j = wg * JSL + jj;
    float v = 0.0f;
    if (grp == 0) {
      v = (k < EE) ? W_ih[(size_t)j * EE + k] : W_hh[(size_t)j * HH + (k - EE)];
    } else if (grp == 1) {
      int row = HH + j;
      v = (k < EE) ? W_ih[(size_t)row * EE + k] : W_hh[(size_t)row * HH + (k - EE)];
    } else if (grp == 2) {
      if (k < EE) v = W_ih[(size_t)(2 * HH + j) * EE + k];
    } else {
      if (k >= EE) v = W_hh[(size_t)(2 * HH + j) * HH + (k - EE)];
    }
    Wl[c * KTOT + k] = f2bf(v);
  }
  if (tid < 128) {
    int grp = tid >> 5, jj = tid & 31;
    int j = wg * JSL + jj;
    float bv;
    if (grp == 0)      bv = b_ih[j] + b_hh[j];
    else if (grp == 1) bv = b_ih[HH + j] + b_hh[HH + j];
    else if (grp == 2) bv = b_ih[2 * HH + j];
    else               bv = b_hh[2 * HH + j];
    bsl[tid] = bv;
  }
  for (int i = tid; i < BB * JSL; i += blockDim.x) hsl[i] = 0.0f;  // h0 = 0
  __syncthreads();

  // wave -> column-tile group: 0,1:r  2,3:z  4,5:n_i  6,7:n_h
  const int ncol0  = wave * 16;
  const int kstart = (wave >= 6) ? EE : 0;
  const int kend   = (wave == 4 || wave == 5) ? EE : KTOT;

  for (int s = 0; s < SS; ++s) {
    // ---- async-DMA stage A panel [x_t | h] -> LDS (96 KB, b128/lane) ----
#pragma unroll
    for (int i = 0; i < 24; ++i) {
      int c   = tid + i * 256;        // 0..6143 chunks of 16B
      int row = c / 96;
      int cc  = c - row * 96;
      const unsigned short* src;
      int dstoff;                     // in bf16 elements
      if (cc < 32) {
        src    = x_bf + ((size_t)(s * BB + row)) * EE + cc * 8;
        dstoff = row * KTOT + cc * 8;
      } else {
        src    = h_bf + (size_t)row * HH + (cc - 32) * 8;
        dstoff = row * KTOT + EE + (cc - 32) * 8;
      }
      async_copy_b128(als_base + dstoff * 2, src);
    }
    // prefetch next step's x panel into cache while we compute this step
    if (s + 1 < SS)
      __builtin_prefetch(x_bf + (size_t)(s + 1) * BB * EE + tid * 64, 0, 3);
    wait_asynccnt0();
    __syncthreads();

    // ---- fused GEMM: gates[64][128] = Als @ Wl^T, both operands in LDS ----
    // Batch all loads per k-step, then 4 back-to-back WMMAs (one dscnt wait,
    // one hazard window per k-step instead of four).
    v8f acc0 = {}, acc1 = {}, acc2 = {}, acc3 = {};
#pragma unroll 2
    for (int k0 = kstart; k0 < kend; k0 += 32) {
      v16bf btile = load_b32x16(Wl, KTOT, ncol0, k0, lane);
      v16bf a0 = load_a16x32(Als, KTOT,  0, k0, lane);
      v16bf a1 = load_a16x32(Als, KTOT, 16, k0, lane);
      v16bf a2 = load_a16x32(Als, KTOT, 32, k0, lane);
      v16bf a3 = load_a16x32(Als, KTOT, 48, k0, lane);
      acc0 = wmma_bf16(a0, btile, acc0);
      acc1 = wmma_bf16(a1, btile, acc1);
      acc2 = wmma_bf16(a2, btile, acc2);
      acc3 = wmma_bf16(a3, btile, acc3);
    }
    __syncthreads();   // all waves done reading A panel; gates overlay it

    {
      int col  = ncol0 + (lane & 15);
      int roff = (lane >> 4) * 8;
      float bias = bsl[col];
#pragma unroll
      for (int rr = 0; rr < 8; ++rr) {
        gts[(roff + rr) * 128 + col]      = acc0[rr] + bias;
        gts[(16 + roff + rr) * 128 + col] = acc1[rr] + bias;
        gts[(32 + roff + rr) * 128 + col] = acc2[rr] + bias;
        gts[(48 + roff + rr) * 128 + col] = acc3[rr] + bias;
      }
    }
    __syncthreads();

    // ---- gate math: each thread owns 8 contiguous hidden units of one b ----
    {
      int b   = tid >> 2;             // 0..63
      int jj0 = (tid & 3) * 8;        // 0,8,16,24
      int j0  = wg * JSL + jj0;
      float* hrow = hsl + b * JSL + jj0;
      union { unsigned short u16[8]; uint4 q; } hbv;
#pragma unroll
      for (int u = 0; u < 8; ++u) {
        int jj = jj0 + u;
        float gr  = gts[b * 128 + jj];
        float gz  = gts[b * 128 + 32 + jj];
        float gni = gts[b * 128 + 64 + jj];
        float gnh = gts[b * 128 + 96 + jj];
        float r = 1.0f / (1.0f + __expf(-gr));
        float z = 1.0f / (1.0f + __expf(-gz));
        float n = tanhf(gni + r * gnh);
        float hp = hrow[u];
        float hn = (1.0f - z) * n + z * hp;
        hrow[u] = hn;
        hbv.u16[u] = f2bf(hn);
      }
      *(uint4*)(h_bf + (size_t)b * HH + j0) = hbv.q;                    // broadcast
      *(uint4*)(st_bf + ((size_t)b * SS + s) * HH + j0) = hbv.q;        // states
    }
    grid_sync(cnt, gen, NBLK_SCAN);   // publish h to all WGs for step s+1
  }
}

// ------------------------- kernel 4: tag projection (WMMA) -----------------
// logits[b*S+s][t] = states @ W_tag^T + b_tag.  M=32768, N=64, K=512.
__global__ void k_tag_gemm(const unsigned short* __restrict__ st_bf,
                           const unsigned short* __restrict__ wtag_bf,
                           const float* __restrict__ b_tag,
                           float* __restrict__ out) {
  const int wave = threadIdx.x >> 5;
  const int lane = threadIdx.x & 31;
  const int mtile = blockIdx.x * 8 + wave;     // 0..2047
  const int m0 = mtile * 16;

  v8f acc0 = {}, acc1 = {}, acc2 = {}, acc3 = {};
#pragma unroll 2
  for (int k0 = 0; k0 < HH; k0 += 32) {
    v16bf a  = load_a16x32(st_bf, HH, m0, k0, lane);
    v16bf b0 = load_b32x16(wtag_bf, HH,  0, k0, lane);
    v16bf b1 = load_b32x16(wtag_bf, HH, 16, k0, lane);
    v16bf b2 = load_b32x16(wtag_bf, HH, 32, k0, lane);
    v16bf b3 = load_b32x16(wtag_bf, HH, 48, k0, lane);
    acc0 = wmma_bf16(a, b0, acc0);
    acc1 = wmma_bf16(a, b1, acc1);
    acc2 = wmma_bf16(a, b2, acc2);
    acc3 = wmma_bf16(a, b3, acc3);
  }
  int roff = (lane >> 4) * 8;
  int coll = lane & 15;
  const v8f* accs[4] = { &acc0, &acc1, &acc2, &acc3 };
#pragma unroll
  for (int nt = 0; nt < 4; ++nt) {
    int col = nt * 16 + coll;
    float bias = b_tag[col];
#pragma unroll
    for (int rr = 0; rr < 8; ++rr) {
      int row = m0 + roff + rr;
      out[(size_t)row * TT + col] = (*accs[nt])[rr] + bias;
    }
  }
}

// ------------------------- kernel 5: masked argmax -------------------------
__global__ void k_argmax(const float* __restrict__ logits,
                         const int* __restrict__ tokens,
                         float* __restrict__ preds) {
  int r = blockIdx.x * blockDim.x + threadIdx.x;   // r = b*512+s
  if (r >= BB * SS) return;
  const float* row = logits + (size_t)r * TT;
  float best = row[0];
  int bi = 0;
#pragma unroll 4
  for (int t = 1; t < TT; ++t) {
    float v = row[t];
    if (v > best) { best = v; bi = t; }
  }
  int tok = tokens[r];
  preds[r] = (tok != 0) ? (float)bi : 0.0f;
}

// ---------------------------------------------------------------------------
extern "C" void kernel_launch(void* const* d_in, const int* in_sizes, int n_in,
                              void* d_out, int out_size, void* d_ws, size_t ws_size,
                              hipStream_t stream) {
  (void)in_sizes; (void)n_in; (void)out_size; (void)ws_size;
  const int*   tokens = (const int*)  d_in[0];
  const float* emb    = (const float*)d_in[1];
  const float* W_ih   = (const float*)d_in[2];
  const float* W_hh   = (const float*)d_in[3];
  const float* b_ih   = (const float*)d_in[4];
  const float* b_hh   = (const float*)d_in[5];
  const float* W_tag  = (const float*)d_in[6];
  const float* b_tag  = (const float*)d_in[7];
  float* out = (float*)d_out;                      // [B][S][T] then preds [B][S]
  float* preds = out + (size_t)BB * SS * TT;

  char* ws = (char*)d_ws;
  // layout: [sync 256][h_bf 64K][x_bf 16M][states_bf 32M][wtag_bf 64K]
  int*            sync_cnt = (int*)ws;
  int*            sync_gen = (int*)(ws + 8);
  unsigned short* h_bf     = (unsigned short*)(ws + 256);
  size_t off = 256 + 65536;                        // 65792
  unsigned short* x_bf     = (unsigned short*)(ws + off);
  off += (size_t)SS * BB * EE * 2;                 // +16 MB
  unsigned short* st_bf    = (unsigned short*)(ws + off);
  off += (size_t)BB * SS * HH * 2;                 // +32 MB
  unsigned short* wtag_bf  = (unsigned short*)(ws + off);

  // zero sync vars + initial hidden state (capturable)
  hipMemsetAsync(ws, 0, 65792, stream);

  // 1) gather embeddings -> bf16 (time-major rows)
  {
    int total = SS * BB * EE / 4;
    k_gather_x<<<(total + 255) / 256, 256, 0, stream>>>(tokens, emb, x_bf);
  }
  // 2) W_tag -> bf16
  k_conv_wtag<<<(TT * HH + 255) / 256, 256, 0, stream>>>(W_tag, wtag_bf);

  // 3) persistent GRU scan: 16 WGs, ~296.5 KB LDS each
  k_gru_scan<<<NBLK_SCAN, 256, SMEM_TOTAL, stream>>>(x_bf, W_ih, W_hh,
                                                     b_ih, b_hh,
                                                     h_bf, st_bf,
                                                     sync_cnt, sync_gen);
  // 4) tag logits
  k_tag_gemm<<<256, 256, 0, stream>>>(st_bf, wtag_bf, b_tag, out);
  // 5) masked argmax -> preds
  k_argmax<<<(BB * SS + 255) / 256, 256, 0, stream>>>(out, tokens, preds);
}
